// KANLayer_43284680409604
// MI455X (gfx1250) — matrix operand order repaired and checked by reading, not compiled, not run on hardware
//
#include <hip/hip_runtime.h>

typedef __attribute__((ext_vector_type(16))) _Float16 v16h;
typedef __attribute__((ext_vector_type(8)))  _Float16 v8h;
typedef __attribute__((ext_vector_type(4)))  _Float16 v4h;
typedef __attribute__((ext_vector_type(8)))  float    v8f;

#define IN_DIM   512
#define OUT_DIM  512
#define NB       16
#define KBASIS   (IN_DIM * NB)       // 8192
#define KTOT     (KBASIS + IN_DIM)   // 8704 fused K (basis | silu-base)
#define KT       32                  // K depth per f16 WMMA step
#define MT       64                  // batch rows per block
#define NTHREADS 512                 // 16 waves (wave32)
#define KB_ITERS   (KBASIS / KT)     // 256
#define TOT_ITERS  (KTOT / KT)       // 272
#define TAIL_ITERS (IN_DIM / KT)     // 16
#define ABYTES   (MT * KT * 2)       // 4 KB A stage
#define BUFBYTES (ABYTES + OUT_DIM * KT * 2)  // 36 KB per stage buffer (3x = 108 KB)
#define WS_NEED  ((size_t)OUT_DIM * KTOT * sizeof(_Float16))  // 8.9 MB f16 weights

// ======================= shared helpers =======================

__device__ __forceinline__ void wmma_step(const _Float16* __restrict__ As,
                                          const _Float16* __restrict__ Bs,
                                          int mt, int ntb, int m, int hh,
                                          v8f acc[8]) {
    int arow = mt * 16 + m;
    v8h alo = *(const v8h*)(As + arow * KT + hh * 16);      // K = hh*16 .. +8
    v8h ahi = *(const v8h*)(As + arow * KT + hh * 16 + 8);  // K = hh*16+8 .. +8
    v16h a = __builtin_shufflevector(alo, ahi,
             0,1,2,3,4,5,6,7,8,9,10,11,12,13,14,15);
#pragma unroll
    for (int g = 0; g < 2; ++g) {
        v16h bf[4];
#pragma unroll
        for (int j2 = 0; j2 < 4; ++j2) {
            int jrow = (ntb + g * 4 + j2) * 16 + m;         // N = m, K seg = hh*16
            bf[j2] = *(const v16h*)(Bs + jrow * KT + hh * 16);
        }
#pragma unroll
        for (int j2 = 0; j2 < 4; ++j2)
            acc[g * 4 + j2] = __builtin_amdgcn_wmma_f32_16x16x32_f16(
                false, a, false, bf[j2], (short)0, acc[g * 4 + j2], false, false);
    }
}

__device__ __forceinline__ void basis_fill(_Float16* As, float xreg,
                                           int r_fill, int g_fill,
                                           float hstep, float inv_h) {
    v4h hv;
#pragma unroll
    for (int kk = 0; kk < 4; ++kk) {
        int   kb = (g_fill * 4 + kk) & 15;
        float c  = -2.0f + kb * hstep;
        float z  = (xreg - c) * inv_h;
        hv[kk] = (_Float16)__expf(-z * z);
    }
    *(v4h*)(As + r_fill * KT + g_fill * 4) = hv;
}

__device__ __forceinline__ float silu1(float v) {
    return v * __builtin_amdgcn_rcpf(1.0f + __expf(-v));   // v_rcp_f32, no IEEE div
}

__device__ __forceinline__ void silu_fill(_Float16* As, float4 xv,
                                          int r_fill, int g_fill) {
    v4h hv;
    hv[0] = (_Float16)silu1(xv.x);
    hv[1] = (_Float16)silu1(xv.y);
    hv[2] = (_Float16)silu1(xv.z);
    hv[3] = (_Float16)silu1(xv.w);
    *(v4h*)(As + r_fill * KT + g_fill * 4) = hv;
}

__device__ __forceinline__ void epilogue(float* Cs, v8f acc[8],
                                         const float* gamma, const float* beta,
                                         float* out, int b0, int t,
                                         int mt, int ntb, int m, int hh) {
#pragma unroll
    for (int ti = 0; ti < 8; ++ti) {
        int ncol = (ntb + ti) * 16 + m;
#pragma unroll
        for (int r = 0; r < 8; ++r) {
            int row = mt * 16 + hh * 8 + r;   // C/D layout: M = r + hh*8, N = lane&15
            Cs[row * OUT_DIM + ncol] = acc[ti][r];
        }
    }
    __syncthreads();
    int row = t >> 3;
    int g   = t & 7;
    float s = 0.f, ss = 0.f;
#pragma unroll 8
    for (int c = 0; c < 64; ++c) {
        float v = Cs[row * OUT_DIM + c * 8 + g];
        s += v; ss += v * v;
    }
#pragma unroll
    for (int mask = 1; mask < 8; mask <<= 1) {
        s  += __shfl_xor(s,  mask, 32);
        ss += __shfl_xor(ss, mask, 32);
    }
    float mean = s * (1.0f / OUT_DIM);
    float var  = ss * (1.0f / OUT_DIM) - mean * mean;
    float rstd = rsqrtf(var + 1e-5f);
    float* orow = out + (size_t)(b0 + row) * OUT_DIM;
#pragma unroll 8
    for (int c = 0; c < 64; ++c) {
        int n = c * 8 + g;
        orow[n] = (Cs[row * OUT_DIM + n] - mean) * rstd * gamma[n] + beta[n];
    }
}

// ======================= async path =======================

__global__ __launch_bounds__(256)
void kan_prep_wf16(const float* __restrict__ w, const float* __restrict__ bw,
                   _Float16* __restrict__ wf) {
    int j = blockIdx.x;              // 0..511
    int t = threadIdx.x;             // 256 threads
    const float* src = w + (size_t)j * KBASIS;
    _Float16*    dst = wf + (size_t)j * KTOT;
#pragma unroll
    for (int c = 0; c < 8; ++c) {
        int k = (t + c * 256) * 4;
        float4 v = *(const float4*)(src + k);
        v4h hv;
        hv[0] = (_Float16)v.x; hv[1] = (_Float16)v.y;
        hv[2] = (_Float16)v.z; hv[3] = (_Float16)v.w;
        *(v4h*)(dst + k) = hv;
    }
    if (t < 128) {
        int k = t * 4;
        float4 v = *(const float4*)(bw + (size_t)j * IN_DIM + k);
        v4h hv;
        hv[0] = (_Float16)v.x; hv[1] = (_Float16)v.y;
        hv[2] = (_Float16)v.z; hv[3] = (_Float16)v.w;
        *(v4h*)(dst + KBASIS + k) = hv;
    }
}

// issue one B tile (512 x 32 f16 = 32 KB) as 4 async B128 DMAs per thread
__device__ __forceinline__ void issue_b_async(const _Float16* __restrict__ wf,
                                              int k0, unsigned ldsB, int t) {
#pragma unroll
    for (int c = 0; c < 4; ++c) {
        int idx = t + c * NTHREADS;           // 0..2047 16-byte chunks
        int j   = idx >> 2;
        int kq  = (idx & 3) << 3;             // halves
        unsigned long long ga =
            (unsigned long long)(const void*)(wf + (size_t)j * KTOT + k0 + kq);
        unsigned la = ldsB + (unsigned)((j * KT + kq) * 2);
        asm volatile("global_load_async_to_lds_b128 %0, %1, off"
                     :: "v"(la), "v"(ga) : "memory");
    }
}

__launch_bounds__(NTHREADS)
__global__ void kan_fused_async(const float* __restrict__ x,
                                const _Float16* __restrict__ wf,
                                const float* __restrict__ gamma,
                                const float* __restrict__ beta,
                                float* __restrict__ out)
{
    extern __shared__ char smem[];
    float* Cs = (float*)smem;                // epilogue overlay (128 KB)
    const unsigned lds0 = (unsigned)(unsigned long long)(void*)smem;

    const int t    = threadIdx.x;
    const int lane = t & 31;
    const int w_id = t >> 5;
    const int b0   = blockIdx.x * MT;
    const int mt   = w_id & 3;
    const int ntb  = (w_id >> 2) * 8;

    v8f acc[8];
#pragma unroll
    for (int i = 0; i < 8; ++i) acc[i] = v8f{};

    const int r_fill = t >> 3;
    const int g_fill = t & 7;
    const size_t xbase = (size_t)(b0 + r_fill) * IN_DIM;
    const int   xo    = g_fill >> 2;
    const float hstep = 4.0f / (NB - 1);
    const float inv_h = 1.0f / hstep;
    const int m  = lane & 15;
    const int hh = lane >> 4;

    float  xreg;
    float4 xreg4;

    // depth-2 prologue: DMA tiles 0,1 into bufs 0,1 (ASYNCcnt = 8)
    issue_b_async(wf, 0,  lds0 + 0 * BUFBYTES + ABYTES, t);
    issue_b_async(wf, KT, lds0 + 1 * BUFBYTES + ABYTES, t);
    xreg = x[xbase + xo];

    int p = 0, p2 = 2;   // consume buffer / issue buffer (mod 3)
    for (int it = 0; it < TOT_ITERS; ++it) {
        _Float16* As = (_Float16*)(smem + p * BUFBYTES);
        _Float16* Bs = (_Float16*)(smem + p * BUFBYTES + ABYTES);

        if (it < KB_ITERS) basis_fill(As, xreg, r_fill, g_fill, hstep, inv_h);
        else               silu_fill(As, xreg4, r_fill, g_fill);

        // in-order ASYNCcnt: <=4 means this tile's 4 DMAs have landed,
        // while next tile's 4 stay in flight. Drain fully on the last tile.
        if (it + 1 < TOT_ITERS)
            asm volatile("s_wait_asynccnt 0x4" ::: "memory");
        else
            asm volatile("s_wait_asynccnt 0x0" ::: "memory");
        __syncthreads();

        if (it + 2 < TOT_ITERS)
            issue_b_async(wf, (it + 2) * KT, lds0 + p2 * BUFBYTES + ABYTES, t);
        if (it + 1 < KB_ITERS)
            xreg = x[xbase + (it + 1) * 2 + xo];
        else if (it + 1 < TOT_ITERS)
            xreg4 = *(const float4*)(x + xbase + (it + 1 - KB_ITERS) * KT + g_fill * 4);

        wmma_step(As, Bs, mt, ntb, m, hh, acc);
        p  = (p  == 2) ? 0 : p  + 1;
        p2 = (p2 == 2) ? 0 : p2 + 1;
    }

    __syncthreads();
    epilogue(Cs, acc, gamma, beta, out, b0, t, mt, ntb, m, hh);
}

// ======================= fallback path (no workspace) =======================

__device__ __forceinline__ void load_w_tile(const float* __restrict__ src,
                                            int k0, int stride, int t,
                                            float4 wreg[8]) {
#pragma unroll
    for (int c = 0; c < 8; ++c) {
        int idx = t + c * NTHREADS;
        int j   = idx >> 3;
        int kq  = (idx & 7) << 2;
        wreg[c] = *(const float4*)(src + (size_t)j * stride + k0 + kq);
    }
}

__device__ __forceinline__ void store_w_tile(_Float16* __restrict__ Bs,
                                             const float4 wreg[8], int t) {
#pragma unroll
    for (int c = 0; c < 8; ++c) {
        int idx = t + c * NTHREADS;
        int j   = idx >> 3;
        int kq  = (idx & 7) << 2;
        v4h hv;
        hv[0] = (_Float16)wreg[c].x; hv[1] = (_Float16)wreg[c].y;
        hv[2] = (_Float16)wreg[c].z; hv[3] = (_Float16)wreg[c].w;
        *(v4h*)(Bs + j * KT + kq) = hv;
    }
}

__launch_bounds__(NTHREADS)
__global__ void kan_fused_kernel(const float* __restrict__ x,
                                 const float* __restrict__ w,
                                 const float* __restrict__ bw,
                                 const float* __restrict__ gamma,
                                 const float* __restrict__ beta,
                                 float* __restrict__ out)
{
    extern __shared__ char smem[];
    float* Cs = (float*)smem;

    const int t    = threadIdx.x;
    const int lane = t & 31;
    const int w_id = t >> 5;
    const int b0   = blockIdx.x * MT;
    const int mt   = w_id & 3;
    const int ntb  = (w_id >> 2) * 8;

    v8f acc[8];
#pragma unroll
    for (int i = 0; i < 8; ++i) acc[i] = v8f{};

    const int r_fill = t >> 3;
    const int g_fill = t & 7;
    const size_t xbase = (size_t)(b0 + r_fill) * IN_DIM;
    const int   xo    = g_fill >> 2;
    const float hstep = 4.0f / (NB - 1);
    const float inv_h = 1.0f / hstep;
    const int m  = lane & 15;
    const int hh = lane >> 4;

    float4 wreg[8];
    float  xreg;
    float4 xreg4;

    load_w_tile(w, 0, KBASIS, t, wreg);
    xreg = x[xbase + xo];

    int p = 0;
    for (int it = 0; it < KB_ITERS; ++it) {
        _Float16* As = (_Float16*)(smem + p * BUFBYTES);
        _Float16* Bs = (_Float16*)(smem + p * BUFBYTES + ABYTES);

        store_w_tile(Bs, wreg, t);
        basis_fill(As, xreg, r_fill, g_fill, hstep, inv_h);
        __syncthreads();

        if (it + 1 < KB_ITERS) {
            load_w_tile(w, (it + 1) * KT, KBASIS, t, wreg);
            xreg = x[xbase + (it + 1) * 2 + xo];
        } else {
            load_w_tile(bw, 0, IN_DIM, t, wreg);
            xreg4 = *(const float4*)(x + xbase + g_fill * 4);
        }
        wmma_step(As, Bs, mt, ntb, m, hh, acc);
        p ^= 1;
    }
    for (int it = 0; it < TAIL_ITERS; ++it) {
        _Float16* As = (_Float16*)(smem + p * BUFBYTES);
        _Float16* Bs = (_Float16*)(smem + p * BUFBYTES + ABYTES);

        store_w_tile(Bs, wreg, t);
        silu_fill(As, xreg4, r_fill, g_fill);
        __syncthreads();

        if (it + 1 < TAIL_ITERS) {
            load_w_tile(bw, (it + 1) * KT, IN_DIM, t, wreg);
            xreg4 = *(const float4*)(x + xbase + (it + 1) * KT + g_fill * 4);
        }
        wmma_step(As, Bs, mt, ntb, m, hh, acc);
        p ^= 1;
    }

    __syncthreads();
    epilogue(Cs, acc, gamma, beta, out, b0, t, mt, ntb, m, hh);
}

// ======================= launcher =======================

extern "C" void kernel_launch(void* const* d_in, const int* in_sizes, int n_in,
                              void* d_out, int out_size, void* d_ws, size_t ws_size,
                              hipStream_t stream) {
    const float* x  = (const float*)d_in[0];
    const float* w  = (const float*)d_in[1];
    const float* bw = (const float*)d_in[2];
    const float* g  = (const float*)d_in[3];
    const float* b  = (const float*)d_in[4];
    float* out = (float*)d_out;

    const int batch = in_sizes[0] / IN_DIM;                    // 4096
    const size_t smem = (size_t)MT * OUT_DIM * sizeof(float);  // 128 KB
    dim3 grid(batch / MT);                                     // 64 blocks

    if (ws_size >= WS_NEED) {
        _Float16* wf = (_Float16*)d_ws;
        kan_prep_wf16<<<OUT_DIM, 256, 0, stream>>>(w, bw, wf);
        hipFuncSetAttribute((const void*)kan_fused_async,
                            hipFuncAttributeMaxDynamicSharedMemorySize, (int)smem);
        kan_fused_async<<<grid, NTHREADS, smem, stream>>>(x, wf, g, b, out);
    } else {
        hipFuncSetAttribute((const void*)kan_fused_kernel,
                            hipFuncAttributeMaxDynamicSharedMemorySize, (int)smem);
        kan_fused_kernel<<<grid, NTHREADS, smem, stream>>>(x, w, bw, g, b, out);
    }
}
